// PathWalkModel_23244363006591
// MI455X (gfx1250) — compile-verified, matrix operand order checked by reading, not run on hardware
//
#include <hip/hip_runtime.h>

typedef _Float16 f16;
typedef __attribute__((ext_vector_type(16))) _Float16 v16h;
typedef __attribute__((ext_vector_type(8)))  _Float16 v8h;
typedef __attribute__((ext_vector_type(8)))  float    v8f;

// Problem constants
static constexpr int  Bb = 32, Ll = 32, KN = 36, Ee = 512, Hh = 512;
static constexpr int  R1 = Bb * Ll;            // 1024 (b,l) rows
static constexpr long R2 = (long)R1 * KN;      // 36864 (b,l,k) rows

__device__ __forceinline__ float gelu_f(float x) {
  return 0.5f * x * (1.0f + erff(x * 0.7071067811865475f));
}
__device__ __forceinline__ float sigm_f(float x) { return 1.0f / (1.0f + expf(-x)); }

// ---------------------------------------------------------------------------
// Generic WMMA GEMM:  out[M,N] = op( A[M,Kd] @ W[N,Kd]^T + bias[N] + addsrc[M,N] )
// A, W f16 row-major. Block = 256 threads (8 waves), tile TBM x TBN, BK=32.
// Waves arranged WAVM x WAVN; each wave owns WM x WN 16x16 output tiles ->
// WM*WN v_wmma_f32_16x16x32_f16 per k-step with WM+WN fragment loads.
//   big:   TBM=128 TBN=64  -> 4 WMMA / 8 ds_load_b128 per wave-step
//   small: TBM=32  TBN=128 -> for the M=32 LSTM recurrent GEMMs
// ---------------------------------------------------------------------------
#define BK 32
#define SK 40   // padded LDS row stride (f16 elems); 80B keeps 16B alignment

template <int TBM, int TBN, int WAVM, int WAVN, int WM, int WN,
          bool GELU, typename OutT>
__global__ __launch_bounds__(256) void gemm_wmma(
    const f16* __restrict__ A, int lda,
    const f16* __restrict__ W, int ldw,
    const float* __restrict__ bias,
    const float* __restrict__ addsrc, long ldadd,
    OutT* __restrict__ out, long ldo,
    int M, int N, int Kd)
{
  static_assert(WAVM * WM * 16 == TBM && WAVN * WN * 16 == TBN, "tile");
  static_assert(WAVM * WAVN == 8, "8 waves");

  __shared__ __align__(16) f16 As[TBM * SK];
  __shared__ __align__(16) f16 Ws[TBN * SK];

  const int tid  = threadIdx.x;
  const int lane = tid & 31;
  const int wave = tid >> 5;
  const int bm = blockIdx.y, bn = blockIdx.x;

  const int mt0 = (wave % WAVM) * WM;   // wave's first M tile
  const int nt0 = (wave / WAVM) * WN;   // wave's first N tile

  v8f acc[WM][WN] = {};

  const int half = (lane & 16) ? 8 : 0;  // ISA 16-bit operand K-pattern
  const int r16  = lane & 15;
  const int lrow = tid >> 2;             // 0..63  (global->LDS staging)
  const int lcol = (tid & 3) * 8;        // 0,8,16,24

  union Frag { v16h v; v8h h[2]; };

  for (int k0 = 0; k0 < Kd; k0 += BK) {
    #pragma unroll
    for (int r = lrow; r < TBM; r += 64) {
      int gm = bm * TBM + r;
      uint4 va = make_uint4(0, 0, 0, 0);
      if (gm < M) va = *(const uint4*)(A + (long)gm * lda + k0 + lcol);
      *(uint4*)(&As[r * SK + lcol]) = va;
    }
    #pragma unroll
    for (int r = lrow; r < TBN; r += 64) {
      int gn = bn * TBN + r;
      uint4 vw = make_uint4(0, 0, 0, 0);
      if (gn < N) vw = *(const uint4*)(W + (long)gn * ldw + k0 + lcol);
      *(uint4*)(&Ws[r * SK + lcol]) = vw;
    }
    // Speculative prefetch of the k+2 tiles (global_prefetch_b8 on gfx1250).
    if (k0 + 2 * BK < Kd) {
      int gm = bm * TBM + lrow;
      if (gm < M) __builtin_prefetch(A + (long)gm * lda + k0 + 2 * BK, 0, 1);
      int gn = bn * TBN + lrow;
      if (lrow < TBN && gn < N)
        __builtin_prefetch(W + (long)gn * ldw + k0 + 2 * BK, 0, 1);
    }
    __syncthreads();

    Frag a[WM], b[WN];
    #pragma unroll
    for (int im = 0; im < WM; ++im) {
      const int arow = ((mt0 + im) * 16 + r16) * SK;
      a[im].h[0] = *(const v8h*)(&As[arow + half]);
      a[im].h[1] = *(const v8h*)(&As[arow + 16 + half]);
    }
    #pragma unroll
    for (int in = 0; in < WN; ++in) {
      const int brow = ((nt0 + in) * 16 + r16) * SK;
      b[in].h[0] = *(const v8h*)(&Ws[brow + half]);
      b[in].h[1] = *(const v8h*)(&Ws[brow + 16 + half]);
    }
    #pragma unroll
    for (int im = 0; im < WM; ++im)
      #pragma unroll
      for (int in = 0; in < WN; ++in)
        acc[im][in] = __builtin_amdgcn_wmma_f32_16x16x32_f16(
            false, a[im].v, false, b[in].v, (short)0, acc[im][in], false, false);
    __syncthreads();
  }

  // Epilogue: C/D layout -> VGPR v holds row (v + 8*(lane>=16)), col (lane&15)
  const int rowoff = (lane & 16) ? 8 : 0;
  #pragma unroll
  for (int in = 0; in < WN; ++in) {
    const int gn = bn * TBN + (nt0 + in) * 16 + r16;
    const float bz = bias ? bias[gn] : 0.0f;
    #pragma unroll
    for (int im = 0; im < WM; ++im) {
      #pragma unroll
      for (int v = 0; v < 8; ++v) {
        int gm = bm * TBM + (mt0 + im) * 16 + rowoff + v;
        if (gm < M) {
          float xv = acc[im][in][v] + bz;
          if (addsrc) xv += addsrc[(long)gm * ldadd + gn];
          if (GELU) xv = gelu_f(xv);
          out[(long)gm * ldo + gn] = (OutT)xv;
        }
      }
    }
  }
}

// Tile configs
#define GEMM_BIG   128, 64, 4, 2, 2, 2
#define GEMM_SMALL 32, 128, 2, 4, 1, 2

// ---------------------------------------------------------------------------
// Utility kernels
// ---------------------------------------------------------------------------
__global__ void cast_kernel(const float* __restrict__ in, f16* __restrict__ out, int n) {
  int i = blockIdx.x * 256 + threadIdx.x;
  if (i < n) out[i] = (f16)in[i];
}

__global__ void addvec_kernel(const float* __restrict__ a, const float* __restrict__ b,
                              float* __restrict__ out, int n) {
  int i = blockIdx.x * 256 + threadIdx.x;
  if (i < n) out[i] = a[i] + b[i];
}

__global__ void zero_kernel(unsigned int* __restrict__ p, int nwords) {
  int i = blockIdx.x * 256 + threadIdx.x;
  if (i < nwords) p[i] = 0u;
}

// Build A_ee = [emb[x_edges] | xe[l+1]-xe[l]]  (f16, row 1024) and the left
// half of A_nf = [xe | (ee written later by GEMM)].
__global__ void gather_x_kernel(const int* __restrict__ x, const int* __restrict__ xe_idx,
                                const float* __restrict__ emb,
                                f16* __restrict__ A_ee, f16* __restrict__ A_nf)
{
  const int r = blockIdx.x;           // (b,l)
  const int l = r & (Ll - 1);
  const long e0 = (long)xe_idx[r] * Ee;
  const long c0 = (long)x[r] * Ee;
  const long c1 = (l < Ll - 1) ? (long)x[r + 1] * Ee : 0;
  for (int c = threadIdx.x; c < Ee; c += 256) {
    float cur = emb[c0 + c];
    A_ee[(long)r * 1024 + c]       = (f16)emb[e0 + c];
    A_ee[(long)r * 1024 + Ee + c]  = (f16)((l < Ll - 1) ? (emb[c1 + c] - cur) : 0.0f);
    A_nf[(long)r * 1024 + c]       = (f16)cur;
  }
}

// buf_e row r = [emb[edge[r]] | emb[nodeB[r]] - emb[nodeA[r]]]
__global__ void gather_e_kernel(const int* __restrict__ edge, const int* __restrict__ nodeB,
                                const int* __restrict__ nodeA, const float* __restrict__ emb,
                                f16* __restrict__ out)
{
  const long r = blockIdx.x;
  const long e0 = (long)edge[r]  * Ee;
  const long b0 = (long)nodeB[r] * Ee;
  const long a0 = (long)nodeA[r] * Ee;
  for (int c = threadIdx.x; c < Ee; c += 256) {
    out[r * 1024 + c]      = (f16)emb[e0 + c];
    out[r * 1024 + Ee + c] = (f16)(emb[b0 + c] - emb[a0 + c]);
  }
}

// A_of middle third = emb[outer]
__global__ void gather_oe_kernel(const int* __restrict__ outer, const float* __restrict__ emb,
                                 f16* __restrict__ A_of)
{
  const long r = blockIdx.x;
  const long o0 = (long)outer[r] * Ee;
  for (int c = threadIdx.x; c < Ee; c += 256)
    A_of[r * 1536 + Ee + c] = (f16)emb[o0 + c];
}

// LayerNorm over 512 cols; one block per row. In-place safe.
template <typename OutT>
__global__ void layernorm_kernel(const float* __restrict__ in, const float* __restrict__ g,
                                 const float* __restrict__ b, OutT* __restrict__ out)
{
  __shared__ float sbuf[256];
  const long r = blockIdx.x;
  const int tid = threadIdx.x;
  float x0 = in[r * 512 + tid], x1 = in[r * 512 + 256 + tid];
  sbuf[tid] = x0 + x1; __syncthreads();
  for (int st = 128; st > 0; st >>= 1) { if (tid < st) sbuf[tid] += sbuf[tid + st]; __syncthreads(); }
  const float mean = sbuf[0] * (1.0f / 512.0f); __syncthreads();
  const float d0 = x0 - mean, d1 = x1 - mean;
  sbuf[tid] = d0 * d0 + d1 * d1; __syncthreads();
  for (int st = 128; st > 0; st >>= 1) { if (tid < st) sbuf[tid] += sbuf[tid + st]; __syncthreads(); }
  const float inv = rsqrtf(sbuf[0] * (1.0f / 512.0f) + 1e-5f); __syncthreads();
  out[r * 512 + tid]       = (OutT)(d0 * inv * g[tid] + b[tid]);
  out[r * 512 + 256 + tid] = (OutT)(d1 * inv * g[256 + tid] + b[256 + tid]);
}

// Per-(b,l): score_k = (x_·w2a + a2b) + y_k·w2b, softmax over K=36,
// h_in = gelu(x_sum + sum_k p_k y_k). One block per (b,l).
__global__ void attn_kernel(const float* __restrict__ xq, const float* __restrict__ y,
                            const float* __restrict__ xsum, const float* __restrict__ a2w,
                            const float* __restrict__ a2b, f16* __restrict__ hout)
{
  __shared__ float sc[KN];
  __shared__ float sbuf[256];
  __shared__ float sbase;
  const long r = blockIdx.x;
  const int tid = threadIdx.x, lane = tid & 31, wave = tid >> 5;

  float p = xq[r * 512 + tid] * a2w[tid] + xq[r * 512 + 256 + tid] * a2w[256 + tid];
  sbuf[tid] = p; __syncthreads();
  for (int st = 128; st > 0; st >>= 1) { if (tid < st) sbuf[tid] += sbuf[tid + st]; __syncthreads(); }
  if (tid == 0) sbase = sbuf[0] + a2b[0];
  __syncthreads();

  for (int k = wave; k < KN; k += 8) {
    const float* yr = y + (r * KN + k) * 512;
    float acc = 0.0f;
    for (int c = lane; c < 512; c += 32) acc += yr[c] * a2w[512 + c];
    for (int off = 16; off > 0; off >>= 1) acc += __shfl_down(acc, off);
    if (lane == 0) sc[k] = sbase + acc;
  }
  __syncthreads();
  if (tid == 0) {
    float m = -1e30f;
    for (int k = 0; k < KN; ++k) m = fmaxf(m, sc[k]);
    float s = 0.0f;
    for (int k = 0; k < KN; ++k) { float e = expf(sc[k] - m); sc[k] = e; s += e; }
    float inv = 1.0f / s;
    for (int k = 0; k < KN; ++k) sc[k] *= inv;
  }
  __syncthreads();
  for (int c = tid; c < 512; c += 256) {
    float acc = 0.0f;
    for (int k = 0; k < KN; ++k) acc += sc[k] * y[(r * KN + k) * 512 + c];
    hout[r * 512 + c] = (f16)gelu_f(xsum[r * 512 + c] + acc);
  }
}

// LSTM gate pointwise: g[32,2048] -> c,h update; optional sequence / final write.
__global__ void lstm_point_kernel(const float* __restrict__ g, float* __restrict__ c,
                                  f16* __restrict__ h16,
                                  f16* __restrict__ seqdst, long seq_row_stride,
                                  float* __restrict__ hfinal)
{
  const int i = blockIdx.x * 256 + threadIdx.x;   // 0..16383
  const int b = i >> 9, j = i & 511;
  const float gi = g[b * 2048 + j];
  const float gf = g[b * 2048 + 512 + j];
  const float gg = g[b * 2048 + 1024 + j];
  const float go = g[b * 2048 + 1536 + j];
  const float cn = sigm_f(gf) * c[i] + sigm_f(gi) * tanhf(gg);
  const float h  = sigm_f(go) * tanhf(cn);
  c[i] = cn;
  h16[i] = (f16)h;
  if (seqdst) seqdst[(long)b * seq_row_stride + j] = (f16)h;
  if (hfinal) hfinal[i] = h;
}

// Final head: logit = [hf|hb] @ enc_W^T + enc_b, softmax over C=2.
__global__ void head_kernel(const float* __restrict__ hf, const float* __restrict__ hb,
                            const float* __restrict__ encW, const float* __restrict__ encb,
                            float* __restrict__ out)
{
  __shared__ float s0[256], s1[256];
  const int b = blockIdx.x, tid = threadIdx.x;
  float a0 = 0.0f, a1 = 0.0f;
  for (int j = tid; j < 1024; j += 256) {
    float v = (j < 512) ? hf[b * 512 + j] : hb[b * 512 + j - 512];
    a0 += v * encW[j];
    a1 += v * encW[1024 + j];
  }
  s0[tid] = a0; s1[tid] = a1; __syncthreads();
  for (int st = 128; st > 0; st >>= 1) {
    if (tid < st) { s0[tid] += s0[tid + st]; s1[tid] += s1[tid + st]; }
    __syncthreads();
  }
  if (tid == 0) {
    float l0 = s0[0] + encb[0], l1 = s1[0] + encb[1];
    float m = fmaxf(l0, l1);
    float e0 = expf(l0 - m), e1 = expf(l1 - m), inv = 1.0f / (e0 + e1);
    out[b * 2 + 0] = e0 * inv;
    out[b * 2 + 1] = e1 * inv;
  }
}

// ---------------------------------------------------------------------------
// Host orchestration
// ---------------------------------------------------------------------------
extern "C" void kernel_launch(void* const* d_in, const int* in_sizes, int n_in,
                              void* d_out, int out_size, void* d_ws, size_t ws_size,
                              hipStream_t stream) {
  (void)in_sizes; (void)n_in; (void)out_size; (void)ws_size;

  const int* x       = (const int*)d_in[0];
  const int* x_edges = (const int*)d_in[1];
  const int* prefix  = (const int*)d_in[2];
  const int* edge1   = (const int*)d_in[3];
  const int* outer   = (const int*)d_in[4];
  const int* edge2   = (const int*)d_in[5];
  const int* succ    = (const int*)d_in[6];
  const float* emb   = (const float*)d_in[8];
  const float* ep_W  = (const float*)d_in[9];
  const float* ep_b  = (const float*)d_in[10];
  const float* nf_W  = (const float*)d_in[11];
  const float* nf_b  = (const float*)d_in[12];
  const float* nf_g  = (const float*)d_in[13];
  const float* nf_bt = (const float*)d_in[14];
  const float* of_W  = (const float*)d_in[15];
  const float* of_b  = (const float*)d_in[16];
  const float* of_g  = (const float*)d_in[17];
  const float* of_bt = (const float*)d_in[18];
  const float* a_g   = (const float*)d_in[19];
  const float* a_bt  = (const float*)d_in[20];
  const float* a1_W  = (const float*)d_in[21];
  const float* a1_b  = (const float*)d_in[22];
  const float* a2_W  = (const float*)d_in[23];
  const float* a2_b  = (const float*)d_in[24];
  const float* enc_W = (const float*)d_in[25];
  const float* enc_b = (const float*)d_in[26];
  const float* lstmW[4][4];   // [dir: l0f,l0b,l1f,l1b][Wih,Whh,bih,bhh]
  for (int d = 0; d < 4; ++d)
    for (int j = 0; j < 4; ++j)
      lstmW[d][j] = (const float*)d_in[27 + d * 4 + j];

  // Workspace bump allocator
  char* base = (char*)d_ws;
  size_t off = 0;
  auto alloc = [&](size_t bytes) -> void* {
    void* p = base + off;
    off = (off + bytes + 255) & ~(size_t)255;
    return p;
  };

  f16* wq_ep = (f16*)alloc(512 * 1024 * 2);
  f16* wq_nf = (f16*)alloc(512 * 1024 * 2);
  f16* wq_of = (f16*)alloc(512 * 1536 * 2);
  f16* wq_a1 = (f16*)alloc(512 * 512 * 2);
  f16* wq_ih[4]; f16* wq_hh[4]; float* bihh[4];
  const int din_dir[4] = {512, 512, 1024, 1024};
  for (int d = 0; d < 4; ++d) {
    wq_ih[d] = (f16*)alloc((size_t)2048 * din_dir[d] * 2);
    wq_hh[d] = (f16*)alloc((size_t)2048 * 512 * 2);
    bihh[d]  = (float*)alloc(2048 * 4);
  }
  f16*   A_ee  = (f16*)alloc((size_t)R1 * 1024 * 2);
  f16*   A_nf  = (f16*)alloc((size_t)R1 * 1024 * 2);
  float* xsum  = (float*)alloc((size_t)R1 * 512 * 4);   // gelu(nf) -> LN in place
  f16*   lnx   = (f16*)alloc((size_t)R1 * 512 * 2);
  float* xq    = (float*)alloc((size_t)R1 * 512 * 4);   // x_
  f16*   buf_e = (f16*)alloc((size_t)R2 * 1024 * 2);    // e1/e2 GEMM input (reused)
  f16*   A_of  = (f16*)alloc((size_t)R2 * 1536 * 2);    // [e1 | oe | e2]
  float* big   = (float*)alloc((size_t)R2 * 512 * 4);   // gelu(of) -> LN -> (reused) y
  f16*   lnn   = (f16*)alloc((size_t)R2 * 512 * 2);
  f16*   h_in  = (f16*)alloc((size_t)R1 * 512 * 2);
  float* xp    = (float*)alloc((size_t)R1 * 2048 * 4);
  f16*   out1  = (f16*)alloc((size_t)R1 * 1024 * 2);
  float* gbuf  = (float*)alloc((size_t)32 * 2048 * 4);
  f16*   h16   = (f16*)alloc((size_t)32 * 512 * 2);
  float* cbuf  = (float*)alloc((size_t)32 * 512 * 4);
  float* hfin  = (float*)alloc((size_t)32 * 512 * 4);
  float* hbin  = (float*)alloc((size_t)32 * 512 * 4);

  auto cast = [&](const float* s, f16* dst, int n) {
    cast_kernel<<<(n + 255) / 256, 256, 0, stream>>>(s, dst, n);
  };
  cast(ep_W, wq_ep, 512 * 1024);
  cast(nf_W, wq_nf, 512 * 1024);
  cast(of_W, wq_of, 512 * 1536);
  cast(a1_W, wq_a1, 512 * 512);
  for (int d = 0; d < 4; ++d) {
    cast(lstmW[d][0], wq_ih[d], 2048 * din_dir[d]);
    cast(lstmW[d][1], wq_hh[d], 2048 * 512);
    addvec_kernel<<<8, 256, 0, stream>>>(lstmW[d][2], lstmW[d][3], bihh[d], 2048);
  }

  // ---- x path ----
  gather_x_kernel<<<R1, 256, 0, stream>>>(x, x_edges, emb, A_ee, A_nf);
  // ee = gelu([emb[x_edges]|sub] @ ep_W^T + ep_b) -> right half of A_nf (f16)
  gemm_wmma<GEMM_BIG, true, f16><<<dim3(512 / 64, R1 / 128), 256, 0, stream>>>(
      A_ee, 1024, wq_ep, 1024, ep_b, nullptr, 0, A_nf + 512, 1024, R1, 512, 1024);
  // gelu([xe|ee] @ nf_W^T + nf_b) -> xsum (f32)
  gemm_wmma<GEMM_BIG, true, float><<<dim3(512 / 64, R1 / 128), 256, 0, stream>>>(
      A_nf, 1024, wq_nf, 1024, nf_b, nullptr, 0, xsum, 512, R1, 512, 1024);
  layernorm_kernel<float><<<R1, 256, 0, stream>>>(xsum, nf_g, nf_bt, xsum);  // x_sum
  layernorm_kernel<f16><<<R1, 256, 0, stream>>>(xsum, a_g, a_bt, lnx);
  // x_ = ln(x_sum) @ a1_W^T + a1_b
  gemm_wmma<GEMM_BIG, false, float><<<dim3(512 / 64, R1 / 128), 256, 0, stream>>>(
      lnx, 512, wq_a1, 512, a1_b, nullptr, 0, xq, 512, R1, 512, 512);

  // ---- neighborhood path ----
  gather_e_kernel<<<(int)R2, 256, 0, stream>>>(edge1, outer, prefix, emb, buf_e);
  gemm_wmma<GEMM_BIG, true, f16><<<dim3(512 / 64, (int)(R2 / 128)), 256, 0, stream>>>(
      buf_e, 1024, wq_ep, 1024, ep_b, nullptr, 0, A_of, 1536, (int)R2, 512, 1024);
  gather_oe_kernel<<<(int)R2, 256, 0, stream>>>(outer, emb, A_of);
  gather_e_kernel<<<(int)R2, 256, 0, stream>>>(edge2, succ, outer, emb, buf_e);
  gemm_wmma<GEMM_BIG, true, f16><<<dim3(512 / 64, (int)(R2 / 128)), 256, 0, stream>>>(
      buf_e, 1024, wq_ep, 1024, ep_b, nullptr, 0, A_of + 1024, 1536, (int)R2, 512, 1024);
  // nbhd = ln(gelu([e1|oe|e2] @ of_W^T + of_b))
  gemm_wmma<GEMM_BIG, true, float><<<dim3(512 / 64, (int)(R2 / 128)), 256, 0, stream>>>(
      A_of, 1536, wq_of, 1536, of_b, nullptr, 0, big, 512, (int)R2, 512, 1536);
  layernorm_kernel<float><<<(int)R2, 256, 0, stream>>>(big, of_g, of_bt, big);
  layernorm_kernel<f16><<<(int)R2, 256, 0, stream>>>(big, a_g, a_bt, lnn);
  // y = ln(nbhd) @ a1_W^T + a1_b  (reuse `big` as y)
  gemm_wmma<GEMM_BIG, false, float><<<dim3(512 / 64, (int)(R2 / 128)), 256, 0, stream>>>(
      lnn, 512, wq_a1, 512, a1_b, nullptr, 0, big, 512, (int)R2, 512, 512);

  // ---- attention mix -> h_in ----
  attn_kernel<<<R1, 256, 0, stream>>>(xq, big, xsum, a2_W, a2_b, h_in);

  // ---- LSTM ----
  auto run_dir = [&](const f16* inA, int din, int d, bool rev,
                     f16* seq_base, int half, float* hfinal) {
    // xp = inA @ Wih^T + (bih+bhh)
    gemm_wmma<GEMM_BIG, false, float><<<dim3(2048 / 64, R1 / 128), 256, 0, stream>>>(
        inA, din, wq_ih[d], din, bihh[d], nullptr, 0, xp, 2048, R1, 2048, din);
    zero_kernel<<<(8192 + 255) / 256, 256, 0, stream>>>((unsigned int*)h16, 8192);
    zero_kernel<<<(16384 + 255) / 256, 256, 0, stream>>>((unsigned int*)cbuf, 16384);
    for (int s = 0; s < Ll; ++s) {
      const int t = rev ? (Ll - 1 - s) : s;
      // g = xp[:,t,:] + h @ Whh^T   (M=32 -> small tile config)
      gemm_wmma<GEMM_SMALL, false, float><<<dim3(2048 / 128, 1), 256, 0, stream>>>(
          h16, 512, wq_hh[d], 512, nullptr, xp + (long)t * 2048, (long)Ll * 2048,
          gbuf, 2048, 32, 2048, 512);
      f16* sd = seq_base ? (seq_base + (long)t * 1024 + half * 512) : nullptr;
      lstm_point_kernel<<<64, 256, 0, stream>>>(gbuf, cbuf, h16, sd,
                                                (long)Ll * 1024, hfinal);
    }
  };

  run_dir(h_in, 512, 0, false, out1, 0, nullptr);   // l0f -> out1[:, :, 0:512]
  run_dir(h_in, 512, 1, true,  out1, 1, nullptr);   // l0b -> out1[:, :, 512:1024]
  run_dir(out1, 1024, 2, false, nullptr, 0, hfin);  // l1f -> final hf
  run_dir(out1, 1024, 3, true,  nullptr, 0, hbin);  // l1b -> final hb

  // ---- head ----
  head_kernel<<<Bb, 256, 0, stream>>>(hfin, hbin, enc_W, enc_b, (float*)d_out);
}
